// RelPositionMultiheadAttentionWeights_18992345383420
// MI455X (gfx1250) — matrix operand order repaired and checked by reading, not compiled, not run on hardware
//
#include <hip/hip_runtime.h>
#include <cstddef>

// ---------------------------------------------------------------------------
// RelPositionMultiheadAttentionWeights for MI455X (gfx1250, wave32, WMMA).
// S=1024 B=8 E=512 H=8 QHD=32 PHD=4 PDIM=192 IN_PROJ=544 (padded to 640)
// bf16 WMMA for both GEMMs; pos-scores folded into WMMA C operand;
// pe window staged in LDS; per-row-tile softmax in LDS.
// ---------------------------------------------------------------------------

typedef __attribute__((ext_vector_type(16))) __bf16 v16bf;
typedef __attribute__((ext_vector_type(8)))  __bf16 v8bf;
typedef __attribute__((ext_vector_type(8)))  float  v8f;

#define S_LEN 1024
#define BATCH 8
#define EDIM  512
#define HEADS 8
#define QHD   32
#define PHD   4
#define PDIM  192
#define INPROJ 544      // real columns
#define NPTILE 40       // padded column tiles (40*16 = 640)

// ---- f32 -> bf16 (round to nearest even) ----
__device__ __forceinline__ unsigned short f2bf(float f) {
    union { float f; unsigned int u; } v; v.f = f;
    unsigned int u = v.u;
    unsigned int r = u + 0x7FFFu + ((u >> 16) & 1u);
    return (unsigned short)(r >> 16);
}

// A-matrix per-lane load: lanes 0-15 get K{0..7,16..23}, lanes 16-31 K{8..15,24..31}
__device__ __forceinline__ v16bf load_a(const unsigned short* p, int grp) {
    const int c0 = grp * 8;
    const int c1 = 16 + grp * 8;
    v8bf lo = *reinterpret_cast<const v8bf*>(p + c0);
    v8bf hi = *reinterpret_cast<const v8bf*>(p + c1);
    v16bf r;
#pragma unroll
    for (int i = 0; i < 8; ++i) { r[i] = lo[i]; r[i + 8] = hi[i]; }
    return r;
}

// B-matrix per-lane load: lanes 0-15 get K{0..7,16..23}, lanes 16-31 K{8..15,24..31}
__device__ __forceinline__ v16bf load_b(const unsigned short* p, int grp) {
    const int c0 = grp * 8;
    v8bf lo = *reinterpret_cast<const v8bf*>(p + c0);
    v8bf hi = *reinterpret_cast<const v8bf*>(p + c0 + 16);
    v16bf r;
#pragma unroll
    for (int i = 0; i < 8; ++i) { r[i] = lo[i]; r[i + 8] = hi[i]; }
    return r;
}

__device__ __forceinline__ v8f wmma_bf16(v16bf a, v16bf b, v8f c) {
    return __builtin_amdgcn_wmma_f32_16x16x32_bf16(false, a, false, b,
                                                   (short)0, c, false, false);
}

// ---------------------------------------------------------------------------
// Kernel: f32 -> bf16 convert
// ---------------------------------------------------------------------------
__global__ __launch_bounds__(256) void k_f32_to_bf16(const float* __restrict__ in,
                                                     unsigned short* __restrict__ out,
                                                     int n) {
    int i = blockIdx.x * 256 + threadIdx.x;
    if (i < n) out[i] = f2bf(in[i]);
}

// ---------------------------------------------------------------------------
// Kernel: pe[n][c] = sum_k pos_emb[n][k] * linear_pos_w[c][k]   (2047 x 32, K=192)
// ---------------------------------------------------------------------------
__global__ __launch_bounds__(256) void k_pe_project(const float* __restrict__ pos_emb,
                                                    const float* __restrict__ lw,
                                                    float* __restrict__ pe) {
    int id = blockIdx.x * 256 + threadIdx.x;
    if (id >= (2 * S_LEN - 1) * 32) return;
    int n = id >> 5;
    int c = id & 31;
    const float* pr = pos_emb + n * PDIM;
    const float* wr = lw + c * PDIM;
    float s = 0.f;
#pragma unroll 4
    for (int k = 0; k < PDIM; ++k) s += pr[k] * wr[k];
    pe[id] = s;
}

// ---------------------------------------------------------------------------
// Kernel: in_proj GEMM (8192 x 640(padded), K=512) via bf16 WMMA.
// Uniform control flow: every wave owns exactly 5 column tiles; K-loop is
// double-buffered so loads of step k+32 overlap the 5 WMMAs of step k.
// Scatters: q,k -> bf16 [b][h][s][32]; p -> f32 [b][h][s][4]; bias added.
// ---------------------------------------------------------------------------
__global__ __launch_bounds__(256) void k_inproj(const unsigned short* __restrict__ xb,
                                                const unsigned short* __restrict__ wb,
                                                const float* __restrict__ bias,
                                                unsigned short* __restrict__ qb,
                                                unsigned short* __restrict__ kb,
                                                float* __restrict__ pb) {
    const int mtile = blockIdx.x;            // 512 row tiles of 16
    const int tid = threadIdx.x;
    const int wid = tid >> 5;                // 8 waves
    const int lane = tid & 31;
    const int lanelo = lane & 15;
    const int grp = lane >> 4;

    const unsigned short* aptr = xb + (mtile * 16 + lanelo) * EDIM;

    const unsigned short* wp[5];
#pragma unroll
    for (int i = 0; i < 5; ++i)
        wp[i] = wb + ((wid + 8 * i) * 16 + lanelo) * EDIM;

    v8f acc[5];
#pragma unroll
    for (int i = 0; i < 5; ++i) { v8f z = {}; acc[i] = z; }

    // preload K-step 0
    v16bf A = load_a(aptr, grp);
    v16bf Bt[5];
#pragma unroll
    for (int i = 0; i < 5; ++i) Bt[i] = load_b(wp[i], grp);

    for (int kk = 32; kk < EDIM; kk += 32) {
        v16bf An = load_a(aptr + kk, grp);
        v16bf Bn[5];
#pragma unroll
        for (int i = 0; i < 5; ++i) Bn[i] = load_b(wp[i] + kk, grp);
#pragma unroll
        for (int i = 0; i < 5; ++i) acc[i] = wmma_bf16(A, Bt[i], acc[i]);
        A = An;
#pragma unroll
        for (int i = 0; i < 5; ++i) Bt[i] = Bn[i];
    }
#pragma unroll
    for (int i = 0; i < 5; ++i) acc[i] = wmma_bf16(A, Bt[i], acc[i]);

    // epilogue: add bias, scatter into q/k (bf16) and p (f32); padded cols dropped
#pragma unroll
    for (int i = 0; i < 5; ++i) {
        int o = (wid + 8 * i) * 16 + lanelo;   // column 0..639
        float bcol = (o < INPROJ) ? bias[o] : 0.f;
#pragma unroll
        for (int r = 0; r < 8; ++r) {
            int m = mtile * 16 + r + grp * 8;   // row in [0,8192)
            int s = m >> 3;                     // seq index
            int b = m & 7;                      // batch index
            float val = acc[i][r] + bcol;
            if (o < 256) {
                int h = o >> 5, d = o & 31;
                qb[(((b * 8 + h) * S_LEN) + s) * 32 + d] = f2bf(val);
            } else if (o < 512) {
                int oo = o - 256;
                int h = oo >> 5, d = oo & 31;
                kb[(((b * 8 + h) * S_LEN) + s) * 32 + d] = f2bf(val);
            } else if (o < INPROJ) {
                int oo = o - 512;
                int h = oo >> 2, d = oo & 3;
                pb[(((b * 8 + h) * S_LEN) + s) * 4 + d] = val;
            }
        }
    }
}

// ---------------------------------------------------------------------------
// Kernel: fused scores (content WMMA + pos via C operand) + mask + softmax.
// grid = (64 row-tiles, 64 hb pairs), 256 threads (8 waves; wave w owns 128 cols).
// Dynamic LDS: scores[16][1024] + pe window[1040 float4] + red[256] + rowv[32].
// ---------------------------------------------------------------------------
__global__ __launch_bounds__(256) void k_attn_softmax(const unsigned short* __restrict__ qb,
                                                      const unsigned short* __restrict__ kb,
                                                      const float* __restrict__ pb,
                                                      const float* __restrict__ pe,
                                                      const unsigned char* __restrict__ mask,
                                                      float* __restrict__ out) {
    extern __shared__ float smem[];
    float* scores = smem;                   // 16*1024
    float* pe_s   = smem + 16 * 1024;       // 1040 float4 (4160 floats)
    float* red    = pe_s + 4 * 1040;        // 256
    float* rowv   = red + 256;              // 32: [0..15] max, [16..31] 1/sum

    const int mtile = blockIdx.x;
    const int hb = blockIdx.y;
    const int h = hb >> 3;
    const int b = hb & 7;
    const int tid = threadIdx.x;
    const int wid = tid >> 5;
    const int lane = tid & 31;
    const int lanelo = lane & 15;
    const int grp = lane >> 4;
    const int m0 = mtile * 16;

    const int bh = b * 8 + h;
    const unsigned short* qbase = qb + ((size_t)bh * S_LEN + m0) * 32;
    const unsigned short* kbase = kb + (size_t)bh * S_LEN * 32;

    // stage pe window: rows pidx = (1008 - m0) + i, i in [0,1039); lds idx i
    // maps to 15 - local_row + col in the score computation.
    {
        const float* peg = pe + (1008 - m0) * 32 + h * 4;
        for (int i = tid; i < 1039; i += 256) {
            float4 v = *reinterpret_cast<const float4*>(peg + i * 32);
            *reinterpret_cast<float4*>(pe_s + i * 4) = v;
        }
    }

    // A tile: q rows m0..m0+15
    v16bf A = load_a(qbase + lanelo * 32, grp);

    // p vectors for the 8 rows this lane owns in C/D layout
    float4 pv[8];
#pragma unroll
    for (int r = 0; r < 8; ++r) {
        int m = m0 + r + grp * 8;
        pv[r] = *reinterpret_cast<const float4*>(pb + ((size_t)bh * S_LEN + m) * 4);
    }

    __syncthreads();

#pragma unroll
    for (int t = 0; t < 8; ++t) {
        int n0 = wid * 128 + t * 16;
        int ncol = n0 + lanelo;
        v16bf Bm = load_b(kbase + ncol * 32, grp);

        // C = pos scores (relative shift folded into LDS index)
        v8f c;
#pragma unroll
        for (int r = 0; r < 8; ++r) {
            int rr = r + grp * 8;                // local row 0..15
            float4 pe4 = *reinterpret_cast<const float4*>(pe_s + (15 - rr + ncol) * 4);
            c[r] = pv[r].x * pe4.x + pv[r].y * pe4.y +
                   pv[r].z * pe4.z + pv[r].w * pe4.w;
        }

        v8f acc = wmma_bf16(A, Bm, c);

        bool mk = mask[b * S_LEN + ncol] != 0;
#pragma unroll
        for (int r = 0; r < 8; ++r) {
            float v = acc[r];
            if (mk) v = -1000.0f;
            scores[(r + grp * 8) * 1024 + ncol] = v;
        }
    }
    __syncthreads();

    // ---- softmax over each of 16 rows (16 threads per row, 64 cols each) ----
    const int row = tid >> 4;
    const int sub = tid & 15;
    float* srow = scores + row * 1024 + sub * 64;

    float pmax = -3.0e38f;
#pragma unroll 8
    for (int c = 0; c < 64; ++c) pmax = fmaxf(pmax, srow[c]);
    red[tid] = pmax;
    __syncthreads();
    if (tid < 16) {
        float mx = red[tid * 16];
#pragma unroll
        for (int i = 1; i < 16; ++i) mx = fmaxf(mx, red[tid * 16 + i]);
        rowv[tid] = mx;
    }
    __syncthreads();

    float mx = rowv[row];
    float psum = 0.f;
#pragma unroll 8
    for (int c = 0; c < 64; ++c) {
        float e = __expf(srow[c] - mx);
        srow[c] = e;
        psum += e;
    }
    red[tid] = psum;
    __syncthreads();
    if (tid < 16) {
        float s = 0.f;
#pragma unroll
        for (int i = 0; i < 16; ++i) s += red[tid * 16 + i];
        rowv[16 + tid] = 1.0f / s;
    }
    __syncthreads();

    // ---- coalesced float4 store: out[(h*8+b), m0+r, :] ----
    float* obase = out + (((size_t)(h * 8 + b) * S_LEN + m0) * S_LEN);
#pragma unroll
    for (int r = 0; r < 16; ++r) {
        float inv = rowv[16 + r];
        float4 v4 = *reinterpret_cast<const float4*>(scores + r * 1024 + tid * 4);
        v4.x *= inv; v4.y *= inv; v4.z *= inv; v4.w *= inv;
        *reinterpret_cast<float4*>(obase + (size_t)r * S_LEN + tid * 4) = v4;
    }
}

// ---------------------------------------------------------------------------
// Launch
// ---------------------------------------------------------------------------
extern "C" void kernel_launch(void* const* d_in, const int* in_sizes, int n_in,
                              void* d_out, int out_size, void* d_ws, size_t ws_size,
                              hipStream_t stream) {
    (void)in_sizes; (void)n_in; (void)out_size; (void)ws_size;

    const float* x        = (const float*)d_in[0];   // (S,B,E)
    const float* pos_emb  = (const float*)d_in[1];   // (1,2S-1,192)
    const unsigned char* mask = (const unsigned char*)d_in[2]; // (B,S) bool
    const float* in_w     = (const float*)d_in[3];   // (544,512)
    const float* in_b     = (const float*)d_in[4];   // (544,)
    const float* lpos_w   = (const float*)d_in[5];   // (32,192)
    float* out            = (float*)d_out;           // (H,B,S,S)

    char* ws = (char*)d_ws;
    unsigned short* xb = (unsigned short*)(ws);                 // 8,388,608 B
    unsigned short* wb = (unsigned short*)(ws + 8388608);       //   655,360 B (640x512 bf16, padded)
    unsigned short* qb = (unsigned short*)(ws + 9043968);       // 4,194,304 B
    unsigned short* kb = (unsigned short*)(ws + 13238272);      // 4,194,304 B
    float*          pb = (float*)(ws + 17432576);               // 1,048,576 B
    float*          pe = (float*)(ws + 18481152);               //   262,016 B

    // 1) convert x and in_proj_w to bf16 (padded weight tail is never consumed)
    {
        int n = S_LEN * BATCH * EDIM;                 // 4,194,304
        k_f32_to_bf16<<<(n + 255) / 256, 256, 0, stream>>>(x, xb, n);
    }
    {
        int n = INPROJ * EDIM;                        // 278,528
        k_f32_to_bf16<<<(n + 255) / 256, 256, 0, stream>>>(in_w, wb, n);
    }

    // 2) pe projection
    {
        int n = (2 * S_LEN - 1) * 32;                 // 65,504
        k_pe_project<<<(n + 255) / 256, 256, 0, stream>>>(pos_emb, lpos_w, pe);
    }

    // 3) in_proj GEMM (WMMA, uniform 5 tiles/wave, double-buffered K loop)
    k_inproj<<<512, 256, 0, stream>>>(xb, wb, in_b, qb, kb, pb);

    // 4) fused scores + pos + mask + softmax (WMMA)
    size_t smem = (size_t)(16 * 1024 + 4 * 1040 + 256 + 32) * sizeof(float); // 83,328 B
    k_attn_softmax<<<dim3(64, 64), 256, smem, stream>>>(qb, kb, pb, pe, mask, out);
}
